// AdditiveCoupling_66915590471963
// MI455X (gfx1250) — matrix operand order repaired and checked by reading, not compiled, not run on hardware
//
#include <hip/hip_runtime.h>

typedef __bf16 bf16_t;
typedef __attribute__((ext_vector_type(16))) __bf16 v16bf;
typedef __attribute__((ext_vector_type(8)))  __bf16 v8bf;
typedef __attribute__((ext_vector_type(8)))  float  v8f;
typedef __attribute__((ext_vector_type(4)))  float  v4f;

#define USE_ASYNC_COPY 1
#define LDS_AS __attribute__((address_space(3)))

constexpr int BM = 128;
constexpr int BN = 128;
constexpr int BK = 32;
constexpr int LDSP = 40;   // padded LDS row stride (elements): 80B keeps 16B align, spreads banks

// ---------------------------------------------------------------------------
// Prep: gather odd columns of x into a bf16 activation buffer; copy log_det_J
// ---------------------------------------------------------------------------
__global__ __launch_bounds__(256) void prep_kernel(const float* __restrict__ x,
                                                   const float* __restrict__ ldj,
                                                   bf16_t* __restrict__ act0,
                                                   float* __restrict__ out_ldj) {
    int i = blockIdx.x * 256 + threadIdx.x;      // over 8192*512
    int b = i >> 9;
    int j = i & 511;
    act0[i] = (bf16_t)x[(size_t)b * 1024 + 2 * j + 1];
    if (i == 0) out_ldj[0] = ldj[0];
}

// ---------------------------------------------------------------------------
// Tiled GEMM: C = A(bf16, MxK) * W(fp32->bf16, KxN) + bias
//   COUPLE=false: ReLU, store bf16 activations (MxN)
//   COUPLE=true : out[m,2n] = x[m,2n] + v ; out[m,2n+1] = x[m,2n+1]
// 256 threads = 8 waves (2x4); each wave: 64x32 output = 4x2 WMMA 16x16 tiles
// A tile: async global->LDS copy (ASYNCcnt).  B tile: manual cvt+transpose.
// ---------------------------------------------------------------------------
template<bool COUPLE>
__global__ __launch_bounds__(256) void gemm_bias_act(
    const bf16_t* __restrict__ A,
    const float*  __restrict__ W,
    const float*  __restrict__ bias,
    bf16_t* __restrict__ outb,
    float*  __restrict__ outf,
    const float* __restrict__ x,
    int M, int N, int K)
{
    __shared__ bf16_t sA[BM * LDSP];
    __shared__ bf16_t sB[BN * LDSP];

    const int tid  = threadIdx.x;
    const int lane = tid & 31;
    const int wid  = tid >> 5;
    const int lrow = lane & 15;
    const bool hiL = lane >= 16;

    const int bm0 = blockIdx.y * BM;
    const int bn0 = blockIdx.x * BN;

    const int wm = (wid & 1) * 64;    // wave M offset in block tile
    const int wn = (wid >> 1) * 32;   // wave N offset in block tile

    v8f acc[4][2];
    for (int i = 0; i < 4; ++i)
        for (int j = 0; j < 2; ++j)
            for (int r = 0; r < 8; ++r) acc[i][j][r] = 0.0f;

    // A-tile staging: each thread moves 16 contiguous bf16 (32B = 2 x b128)
    const int arow  = tid >> 1;        // 0..127
    const int ahalf = (tid & 1) * 16;  // k element offset
    // B-tile staging: each thread reads 16 fp32 along N from one k-row,
    // converts to bf16, writes transposed [N][K]
    const int bk  = tid & 31;
    const int bnc = (tid >> 5) * 16;

    // 32-bit LDS byte address of this thread's A-tile slot (for async copy)
    const unsigned lds_a_addr =
        (unsigned)(uintptr_t)(LDS_AS bf16_t*)&sA[arow * LDSP + ahalf];

    const int ktiles = K / BK;
    for (int kt = 0; kt < ktiles; ++kt) {
        const int k0 = kt * BK;
        {   // A tile -> LDS (row major, K contiguous)
            const bf16_t* ag = A + (size_t)(bm0 + arow) * K + k0 + ahalf;
#if USE_ASYNC_COPY
            // async DMA into LDS; imm offset applies to both global + LDS addr
            asm volatile(
                "global_load_async_to_lds_b128 %0, %1, off\n\t"
                "global_load_async_to_lds_b128 %0, %1, off offset:16"
                :: "v"(lds_a_addr), "v"(ag) : "memory");
#else
            v8bf t0 = *(const v8bf*)(ag);
            v8bf t1 = *(const v8bf*)(ag + 8);
            *(v8bf*)(&sA[arow * LDSP + ahalf])     = t0;
            *(v8bf*)(&sA[arow * LDSP + ahalf + 8]) = t1;
#endif
        }
        {   // W tile -> bf16, transposed into LDS [N][K]
            const float* wg = W + (size_t)(k0 + bk) * N + bn0 + bnc;
            float tmp[16];
            #pragma unroll
            for (int q = 0; q < 4; ++q) {
                v4f t = *(const v4f*)(wg + q * 4);
                tmp[q*4+0] = t[0]; tmp[q*4+1] = t[1];
                tmp[q*4+2] = t[2]; tmp[q*4+3] = t[3];
            }
            #pragma unroll
            for (int q = 0; q < 16; ++q)
                sB[(bnc + q) * LDSP + bk] = (bf16_t)tmp[q];
        }
#if USE_ASYNC_COPY
        asm volatile("s_wait_asynccnt 0" ::: "memory");
#endif
        __syncthreads();

        // prefetch next K-tile into cache (global_prefetch_b8 on gfx1250)
        if (kt + 1 < ktiles) {
            __builtin_prefetch(A + (size_t)(bm0 + arow) * K + k0 + BK + ahalf, 0, 1);
            __builtin_prefetch(W + (size_t)(k0 + BK + bk) * N + bn0 + bnc, 0, 1);
        }

        // B fragments: lane<16 -> K0..15, lane>=16 -> K16..31 (contiguous in LDS row)
        v16bf bfrag[2];
        #pragma unroll
        for (int sn = 0; sn < 2; ++sn) {
            const bf16_t* p = &sB[(wn + sn * 16 + lrow) * LDSP + (hiL ? 16 : 0)];
            v8bf lo = *(const v8bf*)(p);
            v8bf h8 = *(const v8bf*)(p + 8);
            bfrag[sn] = __builtin_shufflevector(lo, h8,
                0,1,2,3,4,5,6,7,8,9,10,11,12,13,14,15);
        }
        // A fragments: lane<16 -> K0..15, lane>=16 -> K16..31
        #pragma unroll
        for (int sm = 0; sm < 4; ++sm) {
            const bf16_t* p = &sA[(wm + sm * 16 + lrow) * LDSP + (hiL ? 16 : 0)];
            v8bf lo = *(const v8bf*)(p);
            v8bf h8 = *(const v8bf*)(p + 8);
            v16bf afrag = __builtin_shufflevector(lo, h8,
                0,1,2,3,4,5,6,7,8,9,10,11,12,13,14,15);
            #pragma unroll
            for (int sn = 0; sn < 2; ++sn) {
                acc[sm][sn] = __builtin_amdgcn_wmma_f32_16x16x32_bf16(
                    false, afrag, false, bfrag[sn],
                    (short)0, acc[sm][sn], false, false);
            }
        }
        __syncthreads();
    }

    // Epilogue. C/D layout: n = lane&15; VGPR r -> m = r + (lane>=16 ? 8 : 0)
    #pragma unroll
    for (int sn = 0; sn < 2; ++sn) {
        const int n  = bn0 + wn + sn * 16 + lrow;
        const float bv = bias[n];
        #pragma unroll
        for (int sm = 0; sm < 4; ++sm) {
            const int mbase = bm0 + wm + sm * 16 + (hiL ? 8 : 0);
            #pragma unroll
            for (int r = 0; r < 8; ++r) {
                float v = acc[sm][sn][r] + bv;
                const int m = mbase + r;
                if (COUPLE) {
                    const size_t idx = (size_t)m * (2 * N) + 2 * n;
                    outf[idx]     = x[idx] + v;
                    outf[idx + 1] = x[idx + 1];
                } else {
                    v = v > 0.f ? v : 0.f;
                    outb[(size_t)m * N + n] = (bf16_t)v;
                }
            }
        }
    }
}

// ---------------------------------------------------------------------------
extern "C" void kernel_launch(void* const* d_in, const int* in_sizes, int n_in,
                              void* d_out, int out_size, void* d_ws, size_t ws_size,
                              hipStream_t stream) {
    const float* x   = (const float*)d_in[0];
    const float* ldj = (const float*)d_in[1];
    const float* Wm[6]; const float* bm[6];
    for (int i = 0; i < 6; ++i) {
        Wm[i] = (const float*)d_in[2 + 2 * i];
        bm[i] = (const float*)d_in[3 + 2 * i];
    }

    const int M = 8192, H = 512, MID = 4096;
    char* ws = (char*)d_ws;
    bf16_t* act0 = (bf16_t*)ws;                                        //  8.4 MB
    bf16_t* bufA = (bf16_t*)(ws + (size_t)M * H * 2);                  // 67.1 MB
    bf16_t* bufB = (bf16_t*)(ws + (size_t)M * H * 2 + (size_t)M * MID * 2);

    float* outf    = (float*)d_out;
    float* out_ldj = outf + (size_t)M * 1024;

    dim3 blk(256);
    prep_kernel<<<dim3((M * H) / 256), blk, 0, stream>>>(x, ldj, act0, out_ldj);

    dim3 gmid(MID / BN, M / BM);
    gemm_bias_act<false><<<gmid, blk, 0, stream>>>(act0, Wm[0], bm[0], bufA, nullptr, nullptr, M, MID, H);
    gemm_bias_act<false><<<gmid, blk, 0, stream>>>(bufA, Wm[1], bm[1], bufB, nullptr, nullptr, M, MID, MID);
    gemm_bias_act<false><<<gmid, blk, 0, stream>>>(bufB, Wm[2], bm[2], bufA, nullptr, nullptr, M, MID, MID);
    gemm_bias_act<false><<<gmid, blk, 0, stream>>>(bufA, Wm[3], bm[3], bufB, nullptr, nullptr, M, MID, MID);
    gemm_bias_act<false><<<gmid, blk, 0, stream>>>(bufB, Wm[4], bm[4], bufA, nullptr, nullptr, M, MID, MID);

    dim3 gout(H / BN, M / BM);
    gemm_bias_act<true><<<gout, blk, 0, stream>>>(bufA, Wm[5], bm[5], nullptr, outf, x, M, H, MID);
}
